// LossFreeMOEDecoder_1108101562791
// MI455X (gfx1250) — matrix operand order repaired and checked by reading, not compiled, not run on hardware
//
#include <hip/hip_runtime.h>
#include <hip/hip_bf16.h>
#include <math.h>

// ---------------------------------------------------------------------------
// LossFreeMOEDecoder forward for gfx1250 (MI455X).
//   tokens N=1024, D=1024, E=64 experts, top-8, FFN 1024->2048->2048->256->768
//
// Layer-split, weight-streaming design:
//  1) gate/route (wave32 per token, wave argmax top-8)
//  2) prefix-sum expert counts -> contiguous routed-row ranges
//  3) compact+stage x -> bf16 A0 [R x 1024]
//  4) four GEMM launches, C = act(A@W + b):
//       block = (expert, 128-col N-chunk, 128-row M-group)
//       8 M-tile accumulators/wave -> each weight chunk streamed ONCE per 128
//       rows (weights ~1.79 GB fp32 read ~1.3x on average => near the
//       23.3 TB/s roofline floor of ~77us). Hidden activations stay bf16 in
//       global scratch (33 MB, L2-resident on the 192 MB L2).
//       LDS staging double-buffered, row stride padded to 40 elems (80 B)
//       -> bank-conflict-free ds_load_b128 fragments; A tile staged with
//       GLOBAL_LOAD_ASYNC_TO_LDS_B128 (ASYNCcnt) when the builtin exists.
//       One barrier per K-chunk, 8 v_wmma_f32_16x16x32_bf16 per barrier.
//  5) combine 8 slots per token (fixed order, no float atomics) + LayerNorm.
// ---------------------------------------------------------------------------

#define E_    64
#define TOPK_ 8
#define N_TOK 1024
#define D_    1024
#define H_    2048
#define BN_   256
#define O_    768
#define R_CAP (N_TOK * TOPK_ + 128)   // routed rows + M-group padding slack
#define LDS_STR 40                    // padded LDS row stride (bf16 elems)

#define AS1 __attribute__((address_space(1)))
#define AS3 __attribute__((address_space(3)))

#if __has_builtin(__builtin_amdgcn_global_load_async_to_lds_b128)
#define HAVE_ASYNC_LDS 1
#else
#define HAVE_ASYNC_LDS 0
#endif

typedef __attribute__((ext_vector_type(16))) __bf16 bf16x16;
typedef __attribute__((ext_vector_type(8)))  float  f32x8;
typedef __attribute__((ext_vector_type(4)))  int    i32x4;

union FragU { bf16x16 f; i32x4 q[2]; };
union PackU { __bf16 h[2]; int i32; };

__device__ __forceinline__ float gelu_exact(float v) {
  return 0.5f * v * (1.0f + erff(v * 0.70710678118654752f));
}

// drain ASYNCcnt (async LDS writes are NOT covered by __syncthreads' DScnt
// wait), then workgroup barrier.
__device__ __forceinline__ void wait_async_and_barrier() {
#if HAVE_ASYNC_LDS
#if __has_builtin(__builtin_amdgcn_s_wait_asynccnt)
  __builtin_amdgcn_s_wait_asynccnt(0);
#else
  asm volatile("s_wait_asynccnt 0x0" ::: "memory");
#endif
#endif
  __syncthreads();
}

// ---------------------------------------------------------------------------
__global__ void init_counts_kernel(int* __restrict__ counts) {
  if (threadIdx.x < E_) counts[threadIdx.x] = 0;
}

// ---------------------------------------------------------------------------
// gating: one wave32 per token; 2 experts/lane; 8 rounds of wave argmax
// (value desc, index asc tie-break == jax top_k); normalized weights.
// ---------------------------------------------------------------------------
__global__ void moe_gate_kernel(const float* __restrict__ x,
                                const float* __restrict__ gw,
                                const float* __restrict__ gbias,
                                int*  __restrict__ counts,
                                int*  __restrict__ tok_list,
                                float* __restrict__ wt_list)
{
  const int gtid = blockIdx.x * blockDim.x + threadIdx.x;
  const int tok  = gtid >> 5;
  const int lane = gtid & 31;
  if (tok >= N_TOK) return;

  const float4* xr = reinterpret_cast<const float4*>(x + (size_t)tok * D_);
  float s[2];
#pragma unroll
  for (int j = 0; j < 2; ++j) {
    const int e = lane + (j << 5);
    const float4* wr = reinterpret_cast<const float4*>(gw + (size_t)e * D_);
    float acc = 0.0f;
    for (int d = 0; d < D_ / 4; ++d) {
      float4 a = xr[d], b = wr[d];
      acc += a.x * b.x + a.y * b.y + a.z * b.z + a.w * b.w;
    }
    s[j] = 1.0f / (1.0f + expf(-acc)) + gbias[e];   // sigmoid + choice bias
  }

  float pv[TOPK_]; int pe[TOPK_]; float sum = 0.0f;
  for (int rsel = 0; rsel < TOPK_; ++rsel) {
    float bv; int be;
    if (s[0] >= s[1]) { bv = s[0]; be = lane; } else { bv = s[1]; be = lane + 32; }
#pragma unroll
    for (int m = 16; m >= 1; m >>= 1) {
      float ov = __shfl_xor(bv, m, 32);
      int   oe = __shfl_xor(be, m, 32);
      if (ov > bv || (ov == bv && oe < be)) { bv = ov; be = oe; }
    }
    pv[rsel] = bv; pe[rsel] = be; sum += bv;
    if (be == lane)           s[0] = -1e30f;
    else if (be == lane + 32) s[1] = -1e30f;
  }

  if (lane == 0) {
    const float inv = 1.0f / (sum + 1e-6f);
    for (int rsel = 0; rsel < TOPK_; ++rsel) {
      const int e = pe[rsel];
      const int pos = atomicAdd(&counts[e], 1);
      tok_list[e * N_TOK + pos] = (tok << 3) | rsel;   // token id + slot
      wt_list[e * N_TOK + pos]  = pv[rsel] * inv;      // normalized weight
    }
  }
}

// ---------------------------------------------------------------------------
// exclusive prefix sum over 64 counters (tiny; one lane)
// ---------------------------------------------------------------------------
__global__ void scan_offsets_kernel(const int* __restrict__ counts,
                                    int* __restrict__ offsets) {
  if (threadIdx.x == 0) {
    int acc = 0;
    for (int e = 0; e < E_; ++e) { offsets[e] = acc; acc += counts[e]; }
    offsets[E_] = acc;
  }
}

// ---------------------------------------------------------------------------
// compact routed-row metadata + stage x fp32 -> A0 bf16 [R x 1024]
// grid: (64 tiles of 16 rows, 64 experts)
// ---------------------------------------------------------------------------
__global__ void compact_stage_kernel(const float* __restrict__ x,
                                     const int* __restrict__ counts,
                                     const int* __restrict__ offsets,
                                     const int* __restrict__ tok_list,
                                     const float* __restrict__ wt_list,
                                     int* __restrict__ r_tok,
                                     int* __restrict__ r_slot,
                                     float* __restrict__ r_wt,
                                     __bf16* __restrict__ A0)
{
  const int e    = blockIdx.y;
  const int cnt  = counts[e];
  const int row0 = blockIdx.x * 16;
  if (row0 >= cnt) return;
  const int rows = min(16, cnt - row0);
  const int base = offsets[e];

  __shared__ int s_tok[16];
  const int tid = threadIdx.x;
  if (tid < 16) {
    int tok = 0;
    if (tid < rows) {
      const int ent = tok_list[e * N_TOK + row0 + tid];
      tok = ent >> 3;
      const int g = base + row0 + tid;
      r_tok[g]  = tok;
      r_slot[g] = ent & 7;
      r_wt[g]   = wt_list[e * N_TOK + row0 + tid];
    }
    s_tok[tid] = tok;
  }
  __syncthreads();

#pragma unroll
  for (int i = 0; i < 16; ++i) {
    const int idx4 = tid + (i << 8);        // 0..4095 float4s (16 rows x 256)
    const int r    = idx4 >> 8;
    const int c4   = (idx4 & 255) << 2;
    if (r < rows) {
      const float4 xv = *reinterpret_cast<const float4*>(
          &x[(size_t)s_tok[r] * D_ + c4]);
      __bf16* dst = &A0[(size_t)(base + row0 + r) * D_ + c4];
      dst[0] = (__bf16)xv.x; dst[1] = (__bf16)xv.y;
      dst[2] = (__bf16)xv.z; dst[3] = (__bf16)xv.w;
    }
  }
}

// ---------------------------------------------------------------------------
// Tiled GEMM: C[rows, NOUT] = act(A[rows, K] @ W[K, NOUT] + b)  per expert.
// block: 256 thr (8 waves, each wave one 16-col N-subtile of a 128-col chunk),
// 8 M-tiles of 16 rows -> 8 accumulators per wave; weight chunk streamed once
// per 128 rows. Double-buffered LDS staging with padded stride (conflict-free
// b128 fragments). A tile is a pure bf16 copy -> async global->LDS when
// available. One barrier per K-chunk; EXEC full at all WMMAs.
// ---------------------------------------------------------------------------
template<int K, int NOUT, bool ACT, bool FINAL>
__global__ __launch_bounds__(256)
void moe_gemm_kernel(const __bf16* __restrict__ Ag,
                     const float* __restrict__ Wg,
                     const float* __restrict__ bias_g,
                     const int* __restrict__ counts,
                     const int* __restrict__ offsets,
                     const int* __restrict__ r_tok,
                     const int* __restrict__ r_slot,
                     const float* __restrict__ r_wt,
                     __bf16* __restrict__ Og,
                     float* __restrict__ outbuf)
{
  const int e      = blockIdx.y;
  const int nb     = blockIdx.x * 128;
  const int cnt    = counts[e];
  const int mstart = blockIdx.z * 128;
  if (mstart >= cnt) return;                 // uniform early-exit
  const int base = offsets[e];

  const float* W    = Wg + (size_t)e * K * NOUT;
  const float* bias = bias_g + (size_t)e * NOUT;

  // padded rows: 40 bf16 = 80 B = 5x16 B -> b128-aligned, and lane banks
  // (n*20 mod 64) are all-distinct across 16 lanes -> conflict-free.
  __shared__ __attribute__((aligned(16))) __bf16 sA[2][128 * LDS_STR]; // 2x10KB
  __shared__ __attribute__((aligned(16))) __bf16 sB[2][128 * LDS_STR]; // 2x10KB

  const int tid  = threadIdx.x;
  const int lane = tid & 31;
  const int wid  = tid >> 5;       // 0..7 -> N-subtile
  const int l15  = lane & 15;
  const int lhf  = lane >> 4;      // 0/1

  auto stage = [&](int kb, int buf) {
    // A tile: pure bf16 copy, 128 rows x 32 k = 512 x 16B; 2 per thread.
#pragma unroll
    for (int i = 0; i < 2; ++i) {
      const int idx = tid + (i << 8);          // 0..511
      const int r   = idx >> 2;
      const int seg = (idx & 3) << 3;          // 0,8,16,24 (bf16 elems)
      const __bf16* gsrc = Ag + (size_t)(base + mstart + r) * K + kb + seg;
      __bf16* ldst = &sA[buf][r * LDS_STR + seg];
#if HAVE_ASYNC_LDS
      AS1 void* gp = (AS1 void*)const_cast<__bf16*>(gsrc);
      AS3 void* lp = (AS3 void*)ldst;
      __builtin_amdgcn_global_load_async_to_lds_b128(
          (AS1 i32x4*)gp, (AS3 i32x4*)lp, 0, 0);
#else
      *reinterpret_cast<i32x4*>(ldst) = *reinterpret_cast<const i32x4*>(gsrc);
#endif
    }
    // W tile: fp32 -> bf16, transposed to [n][k]; pack k-pairs -> b32 stores.
    int* sB32 = reinterpret_cast<int*>(&sB[buf][0]);
#pragma unroll
    for (int i = 0; i < 2; ++i) {
      const int idx = tid + (i << 8);          // 0..511
      const int kp  = idx >> 5;                // 0..15
      const int n4  = (idx & 31) << 2;         // 0..124
      const int k   = kp << 1;
      const float4 wa = *reinterpret_cast<const float4*>(
          &W[(size_t)(kb + k) * NOUT + nb + n4]);
      const float4 wb = *reinterpret_cast<const float4*>(
          &W[(size_t)(kb + k + 1) * NOUT + nb + n4]);
      PackU p;
      p.h[0] = (__bf16)wa.x; p.h[1] = (__bf16)wb.x; sB32[((n4 + 0) * LDS_STR + k) >> 1] = p.i32;
      p.h[0] = (__bf16)wa.y; p.h[1] = (__bf16)wb.y; sB32[((n4 + 1) * LDS_STR + k) >> 1] = p.i32;
      p.h[0] = (__bf16)wa.z; p.h[1] = (__bf16)wb.z; sB32[((n4 + 2) * LDS_STR + k) >> 1] = p.i32;
      p.h[0] = (__bf16)wa.w; p.h[1] = (__bf16)wb.w; sB32[((n4 + 3) * LDS_STR + k) >> 1] = p.i32;
    }
  };

  f32x8 acc[8] = {};

  stage(0, 0);
  wait_async_and_barrier();
  for (int kb = 0; kb < K; kb += 32) {
    const int cur = (kb >> 5) & 1;
    if (kb + 32 < K) {
      stage(kb + 32, cur ^ 1);               // overlap staging with compute
      __builtin_prefetch(&W[(size_t)(kb + 64) * NOUT + nb], 0, 1);
    }
    // B fragment (shared across the 8 M-tiles of this wave's N-subtile)
    FragU b;
    const __bf16* brow = &sB[cur][((wid << 4) + l15) * LDS_STR + (lhf ? 8 : 0)];
    b.q[0] = *reinterpret_cast<const i32x4*>(brow);       // K 0..7  (8..15)
    b.q[1] = *reinterpret_cast<const i32x4*>(brow + 16);  // K 16..23 (24..31)
#pragma unroll
    for (int mt = 0; mt < 8; ++mt) {
      FragU a;
      const __bf16* arow = &sA[cur][((mt << 4) + l15) * LDS_STR + (lhf ? 8 : 0)];
      a.q[0] = *reinterpret_cast<const i32x4*>(arow);      // K +0..7  (8..15)
      a.q[1] = *reinterpret_cast<const i32x4*>(arow + 16); // K +16..23(24..31)
      acc[mt] = __builtin_amdgcn_wmma_f32_16x16x32_bf16(
          false, a.f, false, b.f, (short)0, acc[mt], false, false);
    }
    wait_async_and_barrier();  // all reads of 'cur' done; async writes landed
  }

  // epilogue: C layout -> VGPR r holds rows r (lanes 0-15) / r+8 (lanes 16-31),
  // column fixed per lane -> one bias scalar per lane.
  const int   ncol = nb + (wid << 4) + l15;
  const float bv   = bias[ncol];
#pragma unroll
  for (int mt = 0; mt < 8; ++mt) {
#pragma unroll
    for (int r = 0; r < 8; ++r) {
      const int rloc = mstart + (mt << 4) + (lhf << 3) + r;
      float v = acc[mt][r] + bv;
      if (ACT) v = gelu_exact(v);
      if (rloc < cnt) {
        const int g = base + rloc;
        if (FINAL) {
          outbuf[(size_t)((r_tok[g] << 3) + r_slot[g]) * O_ + ncol] = v * r_wt[g];
        } else {
          Og[(size_t)g * NOUT + ncol] = (__bf16)v;
        }
      }
    }
  }
}

// ---------------------------------------------------------------------------
// combine 8 slots per token (fixed order -> deterministic) + LayerNorm
// ---------------------------------------------------------------------------
__global__ void combine_ln_kernel(const float* __restrict__ outbuf,
                                  const float* __restrict__ ln_w,
                                  const float* __restrict__ ln_b,
                                  float* __restrict__ out)
{
  const int t = blockIdx.x, tid = threadIdx.x;    // 256 thr, O_=768 -> 3 each
  __shared__ float redS[8], redQ[8];
  float v[3]; float ls = 0.f, lq = 0.f;
#pragma unroll
  for (int i = 0; i < 3; ++i) {
    const int o = tid + (i << 8);
    float a = 0.f;
#pragma unroll
    for (int s = 0; s < TOPK_; ++s)
      a += outbuf[(size_t)((t << 3) + s) * O_ + o];
    v[i] = a; ls += a; lq += a * a;
  }
#pragma unroll
  for (int m = 16; m >= 1; m >>= 1) {
    ls += __shfl_xor(ls, m, 32);
    lq += __shfl_xor(lq, m, 32);
  }
  if ((tid & 31) == 0) { redS[tid >> 5] = ls; redQ[tid >> 5] = lq; }
  __syncthreads();
  float ts = 0.f, tq = 0.f;
#pragma unroll
  for (int j = 0; j < 8; ++j) { ts += redS[j]; tq += redQ[j]; }
  const float mu   = ts * (1.0f / O_);
  const float var  = tq * (1.0f / O_) - mu * mu;   // population variance
  const float rstd = rsqrtf(var + 1e-5f);
#pragma unroll
  for (int i = 0; i < 3; ++i) {
    const int o = tid + (i << 8);
    out[(size_t)t * O_ + o] = (v[i] - mu) * rstd * ln_w[o] + ln_b[o];
  }
}

// ---------------------------------------------------------------------------
// launch: init -> gate -> scan -> compact/stage -> L1..L4 GEMMs -> LN
// ws (~112 MB): counts | offsets | tok_list | wt_list | r_tok | r_slot | r_wt
//              | A0 bf16 RxD | h1 bf16 RxH | h2 bf16 RxH | h3 bf16 RxBN
//              | outbuf f32 N*8*O
// ---------------------------------------------------------------------------
extern "C" void kernel_launch(void* const* d_in, const int* in_sizes, int n_in,
                              void* d_out, int out_size, void* d_ws, size_t ws_size,
                              hipStream_t stream)
{
  const float* x      = (const float*)d_in[0];
  const float* gate_w = (const float*)d_in[1];
  const float* gate_b = (const float*)d_in[2];
  const float* w1 = (const float*)d_in[3];
  const float* b1 = (const float*)d_in[4];
  const float* w2 = (const float*)d_in[5];
  const float* b2 = (const float*)d_in[6];
  const float* w3 = (const float*)d_in[7];
  const float* b3 = (const float*)d_in[8];
  const float* w4 = (const float*)d_in[9];
  const float* b4 = (const float*)d_in[10];
  const float* ln_w = (const float*)d_in[11];
  const float* ln_b = (const float*)d_in[12];
  float* out = (float*)d_out;

  char* p = (char*)d_ws;
  auto alloc = [&](size_t bytes) {
    char* r = p; p += (bytes + 255) & ~(size_t)255; return r;
  };
  int*    counts   = (int*)   alloc(E_ * 4);
  int*    offsets  = (int*)   alloc((E_ + 1) * 4);
  int*    tok_list = (int*)   alloc((size_t)E_ * N_TOK * 4);
  float*  wt_list  = (float*) alloc((size_t)E_ * N_TOK * 4);
  int*    r_tok    = (int*)   alloc((size_t)R_CAP * 4);
  int*    r_slot   = (int*)   alloc((size_t)R_CAP * 4);
  float*  r_wt     = (float*) alloc((size_t)R_CAP * 4);
  __bf16* A0       = (__bf16*)alloc((size_t)R_CAP * D_ * 2);
  __bf16* h1       = (__bf16*)alloc((size_t)R_CAP * H_ * 2);
  __bf16* h2       = (__bf16*)alloc((size_t)R_CAP * H_ * 2);
  __bf16* h3       = (__bf16*)alloc((size_t)R_CAP * BN_ * 2);
  float*  outbuf   = (float*) alloc((size_t)N_TOK * TOPK_ * O_ * 4);

  init_counts_kernel<<<1, 64, 0, stream>>>(counts);
  moe_gate_kernel<<<N_TOK / 8, 256, 0, stream>>>(x, gate_w, gate_b,
                                                 counts, tok_list, wt_list);
  scan_offsets_kernel<<<1, 32, 0, stream>>>(counts, offsets);
  compact_stage_kernel<<<dim3(N_TOK / 16, E_), 256, 0, stream>>>(
      x, counts, offsets, tok_list, wt_list, r_tok, r_slot, r_wt, A0);

  const dim3 gH(H_ / 128, E_, N_TOK / 128);    // 16 x 64 x 8
  const dim3 gBN(BN_ / 128, E_, N_TOK / 128);  //  2 x 64 x 8
  const dim3 gO(O_ / 128, E_, N_TOK / 128);    //  6 x 64 x 8
  moe_gemm_kernel<D_, H_,  true,  false><<<gH, 256, 0, stream>>>(
      A0, w1, b1, counts, offsets, r_tok, r_slot, r_wt, h1, (float*)nullptr);
  moe_gemm_kernel<H_, H_,  true,  false><<<gH, 256, 0, stream>>>(
      h1, w2, b2, counts, offsets, r_tok, r_slot, r_wt, h2, (float*)nullptr);
  moe_gemm_kernel<H_, BN_, false, false><<<gBN, 256, 0, stream>>>(
      h2, w3, b3, counts, offsets, r_tok, r_slot, r_wt, h3, (float*)nullptr);
  moe_gemm_kernel<BN_, O_, false, true ><<<gO, 256, 0, stream>>>(
      h3, w4, b4, counts, offsets, r_tok, r_slot, r_wt, (__bf16*)nullptr, outbuf);

  combine_ln_kernel<<<N_TOK, 256, 0, stream>>>(outbuf, ln_w, ln_b, out);

  (void)in_sizes; (void)n_in; (void)out_size; (void)ws_size;
}
